// SpatialEncoder3d_25726854103672
// MI455X (gfx1250) — matrix operand order repaired and checked by reading, not compiled, not run on hardware
//
#include <hip/hip_runtime.h>

typedef _Float16 half_t;
typedef __attribute__((ext_vector_type(16))) _Float16 v16h;
typedef __attribute__((ext_vector_type(8)))  _Float16 v8h;
typedef __attribute__((ext_vector_type(8)))  float    v8f;

#define KDIM 128
#define HDIM 32
#define NN   512
#define NGR  4
#define LDH  136   // padded LDS row stride (halves); keeps 16B alignment, de-conflicts banks

// Load 16 contiguous f16 from LDS (16B-aligned) as a WMMA fragment.
__device__ __forceinline__ v16h lds_load16(const half_t* p) {
    v8h lo = *(const v8h*)(p);
    v8h up = *(const v8h*)(p + 8);
    return __builtin_shufflevector(lo, up, 0,1,2,3,4,5,6,7,8,9,10,11,12,13,14,15);
}

__global__ __launch_bounds__(128)
void spatial_encoder3d_kernel(const float* __restrict__ coord,
                              const int*   __restrict__ ntype,
                              const float* __restrict__ means,
                              const float* __restrict__ stds,
                              const float* __restrict__ mul_w,
                              const float* __restrict__ bias_w,
                              const float* __restrict__ W1,
                              const float* __restrict__ b1,
                              const float* __restrict__ W2,
                              const float* __restrict__ b2,
                              float* __restrict__ out)
{
    __shared__ half_t sW1[KDIM * LDH];      // W1[n][k] as f16, B-fragment friendly (k contiguous)
    __shared__ half_t sW2[HDIM * LDH];      // W2[n][k] as f16
    __shared__ half_t sH[4][16 * LDH];      // per-wave GELU(h) staging, row-major [m][k]
    __shared__ float  sMu[KDIM], sAc[KDIM], sCc[KDIM], sB1[KDIM], sB2[HDIM];

    const int tid = threadIdx.x;

    // Stage weights + per-kernel Gaussian constants once per block.
    for (int idx = tid; idx < KDIM * KDIM; idx += 128)
        sW1[(idx >> 7) * LDH + (idx & 127)] = (half_t)W1[idx];
    for (int idx = tid; idx < HDIM * KDIM; idx += 128)
        sW2[(idx >> 7) * LDH + (idx & 127)] = (half_t)W2[idx];
    if (tid < KDIM) {
        float v  = fabsf(stds[tid]) + 0.01f;
        sMu[tid] = means[tid];
        sAc[tid] = -0.5f / (v * v);
        sCc[tid] = -0.3989422804014327f / v;   // -1/(sqrt(2*pi)*std)
        sB1[tid] = b1[tid];
    }
    if (tid < HDIM) sB2[tid] = b2[tid];
    __syncthreads();

    const int wave  = tid >> 5;      // wave32: 4 waves per block
    const int lane  = tid & 31;
    const int lhalf = lane & 15;     // M row within tile (both half-waves cover M=0..15)
    const int kh    = (lane >> 4) << 3;   // K-phase: 0 for lanes 0-15, 8 for lanes 16-31

    // Row handled by this lane's M position: r = g*N*N + i*N + j
    const int r = (blockIdx.x << 6) + (wave << 4) + lhalf;
    const int g = r >> 18;                 // N*N = 2^18
    const int i = (r >> 9) & (NN - 1);
    const int j = r & (NN - 1);
    const int gi = (g << 9) + i;
    const int gj = (g << 9) + j;

    float dx = coord[gi * 3 + 0] - coord[gj * 3 + 0];
    float dy = coord[gi * 3 + 1] - coord[gj * 3 + 1];
    float dz = coord[gi * 3 + 2] - coord[gj * 3 + 2];
    float d2 = dx * dx + dy * dy + dz * dz;
    float dist = d2 > 0.0f ? sqrtf(d2) : 0.0f;

    int ti = ntype[gi], tj = ntype[gj];
    float mul    = mul_w[ti * 2 + 0]  + mul_w[tj * 2 + 1];
    float bias   = bias_w[ti * 2 + 0] + bias_w[tj * 2 + 1];
    float scaled = mul * dist + bias;

    // -------- GEMM1: h = gk @ W1^T  (A generated in-register, B from LDS) --------
    v8f acc[8] = {};
    #pragma unroll
    for (int s = 0; s < 4; ++s) {
        // A-fragment per ISA 16-bit 16x32 layout:
        //   lane<16: e<8 -> K=32s+e,     e>=8 -> K=32s+16+(e-8)
        //   lane>=16: same with +8 K-phase (kh)
        v16h afrag;
        #pragma unroll
        for (int e = 0; e < 16; ++e) {
            int k   = (s << 5) + kh + ((e < 8) ? e : (e + 8));
            float t = scaled - sMu[k];
            afrag[e] = (half_t)(sCc[k] * __expf(sAc[k] * t * t));
        }
        #pragma unroll
        for (int nt = 0; nt < 8; ++nt) {
            const half_t* bp = &sW1[((nt << 4) + lhalf) * LDH + (s << 5) + kh];
            v16h bfrag = __builtin_shufflevector(*(const v8h*)(bp), *(const v8h*)(bp + 16), 0,1,2,3,4,5,6,7,8,9,10,11,12,13,14,15);
            acc[nt] = __builtin_amdgcn_wmma_f32_16x16x32_f16(
                false, afrag, false, bfrag, (short)0, acc[nt], false, false);
        }
    }

    // -------- bias + exact GELU, stage h as f16 to LDS (C/D layout -> row-major) --------
    half_t* hrow = sH[wave];
    #pragma unroll
    for (int nt = 0; nt < 8; ++nt) {
        #pragma unroll
        for (int v = 0; v < 8; ++v) {
            float x   = acc[nt][v] + sB1[(nt << 4) + lhalf];
            float gel = 0.5f * x * (1.0f + erff(x * 0.7071067811865476f));
            hrow[(v + kh) * LDH + (nt << 4) + lhalf] = (half_t)gel;  // m = v + (hi?8:0)
        }
    }
    // Same-wave LDS ops are in-order (DScnt); reload below observes the stores.

    // -------- GEMM2: out = GELU(h) @ W2^T + b2 --------
    v8f acc2[2] = {};
    #pragma unroll
    for (int s = 0; s < 4; ++s) {
        const half_t* ap = &hrow[lhalf * LDH + (s << 5) + kh];
        v8h lo = *(const v8h*)(ap);          // K = 32s+kh .. +7
        v8h up = *(const v8h*)(ap + 16);     // K = 32s+kh+16 .. +23
        v16h a2 = __builtin_shufflevector(lo, up, 0,1,2,3,4,5,6,7,8,9,10,11,12,13,14,15);
        #pragma unroll
        for (int nt = 0; nt < 2; ++nt) {
            const half_t* bp = &sW2[((nt << 4) + lhalf) * LDH + (s << 5) + kh];
            v16h bfrag = __builtin_shufflevector(*(const v8h*)(bp), *(const v8h*)(bp + 16), 0,1,2,3,4,5,6,7,8,9,10,11,12,13,14,15);
            acc2[nt] = __builtin_amdgcn_wmma_f32_16x16x32_f16(
                false, a2, false, bfrag, (short)0, acc2[nt], false, false);
        }
    }

    // -------- store: out[(row)*32 + h], 64B-contiguous per half-wave --------
    const int rowbase = (blockIdx.x << 6) + (wave << 4);
    #pragma unroll
    for (int nt = 0; nt < 2; ++nt) {
        #pragma unroll
        for (int v = 0; v < 8; ++v) {
            float val = acc2[nt][v] + sB2[(nt << 4) + lhalf];
            int   m   = v + kh;
            out[(size_t)(rowbase + m) * HDIM + (nt << 4) + lhalf] = val;
        }
    }
}

extern "C" void kernel_launch(void* const* d_in, const int* in_sizes, int n_in,
                              void* d_out, int out_size, void* d_ws, size_t ws_size,
                              hipStream_t stream) {
    (void)in_sizes; (void)n_in; (void)out_size; (void)d_ws; (void)ws_size;
    const float* coord  = (const float*)d_in[0];
    const int*   ntype  = (const int*)  d_in[1];
    const float* means  = (const float*)d_in[2];
    const float* stds   = (const float*)d_in[3];
    const float* mul_w  = (const float*)d_in[4];
    const float* bias_w = (const float*)d_in[5];
    const float* W1     = (const float*)d_in[6];
    const float* b1     = (const float*)d_in[7];
    const float* W2     = (const float*)d_in[8];
    const float* b2     = (const float*)d_in[9];
    float* out = (float*)d_out;

    dim3 grid((NGR * NN * NN) / 64);   // 64 rows (4 waves x 16) per block
    dim3 block(128);
    hipLaunchKernelGGL(spatial_encoder3d_kernel, grid, block, 0, stream,
                       coord, ntype, means, stds, mul_w, bias_w, W1, b1, W2, b2, out);
}
